// PatchRNN_68023692034159
// MI455X (gfx1250) — matrix operand order repaired and checked by reading, not compiled, not run on hardware
//
#include <hip/hip_runtime.h>

// Problem constants (match reference)
#define Bq   1024
#define Nn   1024
#define Dd   256
#define Hh   512
#define Kk   16
#define G3H  1536   // 3*H

typedef __bf16 bf16_t;
typedef __attribute__((ext_vector_type(16))) __bf16 v16bf;
typedef __attribute__((ext_vector_type(8)))  __bf16 v8bf;
typedef __attribute__((ext_vector_type(8)))  float  v8f;

union ABFrag { v16bf v; v8bf h[2]; };

// LDS row strides (elements), padded so ds_load_b128 A-fragment loads hit
// distinct banks across lanes (HSTR: stride in dwords mod 64 == 4 -> 16 rows map
// to disjoint 4-dword bank groups).
#define XSTR 272   // 256 cols + 16 pad  -> 136 dwords/row
#define HSTR 520   // 512 cols + 8 pad   -> 260 dwords/row

__device__ __forceinline__ v8f v8f_zero() {
    v8f z;
    #pragma unroll
    for (int i = 0; i < 8; ++i) z[i] = 0.0f;
    return z;
}

__device__ __forceinline__ float fast_sigmoid(float x) {
    return 1.0f / (1.0f + __expf(-x));
}
__device__ __forceinline__ float fast_tanh(float x) {
    // 1 - 2/(e^{2x}+1): well-behaved at +/-inf of the exp
    float e = __expf(2.0f * x);
    return 1.0f - 2.0f / (e + 1.0f);
}

// One-shot f32 -> bf16 conversion of both weight matrices into workspace.
__global__ void cvt_weights_kernel(const float* __restrict__ wih,
                                   const float* __restrict__ whh,
                                   bf16_t* __restrict__ out) {
    const int n1 = G3H * Dd;
    const int n2 = G3H * Hh;
    for (int i = blockIdx.x * blockDim.x + threadIdx.x; i < n1 + n2;
         i += gridDim.x * blockDim.x) {
        float f = (i < n1) ? wih[i] : whh[i - n1];
        out[i] = (bf16_t)f;
    }
}

// A-fragment (16x32 bf16, M x K): lane m = lane&15, half = lane>>4.
// Layout: lane holds K = 16*half + {0..15} (matches B-fragment)
// -> two adjacent 16B chunks per lane.
__device__ __forceinline__ v16bf load_a_frag(const bf16_t* __restrict__ smem,
                                             int row_stride, int m, int half, int kb) {
    ABFrag a;
    const bf16_t* base = smem + m * row_stride + kb + 16 * half;
    a.h[0] = *(const v8bf*)(base);
    a.h[1] = *(const v8bf*)(base + 8);
    return a.v;
}

// B-fragment (32x16 bf16, K x N) with B[k][n] = w[n][k] (row-major weights):
// lane n = lane&15, half = lane>>4; lane holds K = 16*half + {0..15}
// -> one contiguous 32B run from row n of the weight matrix.
__device__ __forceinline__ v16bf load_b_frag(const bf16_t* __restrict__ w,
                                             int row_stride, int row, int half, int kb) {
    return *(const v16bf*)(w + (long long)row * row_stride + kb + 16 * half);
}

// =====================================================================
// Phase 2: fully parallel input projections
//   gi[b*K + k][c] = sum_d tokens[b, idx[b,k], d] * w_ih[c, d] + b_ih[c]
// One block per batch row b (1024 blocks): M-rows of the tile are the K=16
// scan steps. 8 waves x 12 column tiles cover the 1536 output columns.
// =====================================================================
__launch_bounds__(256)
__global__ void gi_gemm_kernel(const float* __restrict__ tokens,
                               const int* __restrict__ indices,
                               const bf16_t* __restrict__ wih,
                               const float* __restrict__ bih,
                               float* __restrict__ gi) {
    __shared__ __align__(32) bf16_t xs[16 * XSTR];

    const int tid  = threadIdx.x;
    const int lane = tid & 31;
    const int wave = tid >> 5;
    const int m    = lane & 15;
    const int half = lane >> 4;
    const int cb   = wave * 192;            // this wave's output-column base
    const long long b = blockIdx.x;

    // Gather the 16 selected token rows for this batch element -> bf16 LDS.
    // Iteration j loads row j; 256 threads cover its 256 columns (coalesced).
    for (int j = 0; j < 16; ++j) {
        int idx = indices[b * Kk + j];
        const float* src = tokens + (b * Nn + idx) * Dd;
        xs[j * XSTR + tid] = (bf16_t)src[tid];
    }
    __syncthreads();

    v8f acc[12];
    #pragma unroll
    for (int t = 0; t < 12; ++t) acc[t] = v8f_zero();

    for (int kb = 0; kb < Dd; kb += 32) {
        v16bf a = load_a_frag(xs, XSTR, m, half, kb);
        #pragma unroll
        for (int t = 0; t < 12; ++t) {
            v16bf bf = load_b_frag(wih, Dd, cb + 16 * t + m, half, kb);
            acc[t] = __builtin_amdgcn_wmma_f32_16x16x32_bf16(
                         false, a, false, bf, (short)0, acc[t], false, false);
        }
    }

    // Bias + store (row stride between M rows is G3H; rows are scan steps).
    #pragma unroll
    for (int t = 0; t < 12; ++t) {
        int col = cb + 16 * t + m;
        float bia = bih[col];
        #pragma unroll
        for (int v = 0; v < 8; ++v) {
            int row = v + 8 * half;                     // = scan step k
            gi[(b * Kk + row) * (long long)G3H + col] = acc[t][v] + bia;
        }
    }
}

// =====================================================================
// Phase 3: serial scan with precomputed gi: per step only gh = h @ w_hh^T.
// 64 blocks x 8 waves; wave owns hidden slice [64w, 64w+64); h stays in
// registers (f32) across all 16 steps, bf16 copy in LDS for A-fragments.
// =====================================================================
__launch_bounds__(256)
__global__ void gru_scan_pre_kernel(const float* __restrict__ gi,
                                    const bf16_t* __restrict__ whh,
                                    const float* __restrict__ bhh,
                                    float* __restrict__ out) {
    __shared__ __align__(32) bf16_t hs[16 * HSTR];

    const int tid  = threadIdx.x;
    const int lane = tid & 31;
    const int wave = tid >> 5;
    const int m    = lane & 15;
    const int half = lane >> 4;
    const int jb   = wave * 64;
    const long long b0 = (long long)blockIdx.x * 16;

    for (int i = tid; i < 16 * HSTR; i += 256) hs[i] = (bf16_t)0.0f;

    // Hoist recurrent biases (gi already contains b_ih).
    float br[4], bz[4], bhn[4];
    #pragma unroll
    for (int t = 0; t < 4; ++t) {
        int c = jb + 16 * t + m;
        br[t]  = bhh[c];
        bz[t]  = bhh[Hh + c];
        bhn[t] = bhh[2 * Hh + c];
    }

    v8f hf[4];
    #pragma unroll
    for (int t = 0; t < 4; ++t) hf[t] = v8f_zero();

    __syncthreads();

    for (int k = 0; k < Kk; ++k) {
        // Seed accumulators with precomputed gi tiles (hot in L2).
        v8f accr[4], accz[4], accni[4], accnh[4];
        #pragma unroll
        for (int t = 0; t < 4; ++t) {
            int cr = jb + 16 * t + m;
            #pragma unroll
            for (int v = 0; v < 8; ++v) {
                long long row = ((b0 + v + 8 * half) * Kk + k) * (long long)G3H;
                accr[t][v]  = gi[row + cr];
                accz[t][v]  = gi[row + Hh + cr];
                accni[t][v] = gi[row + 2 * Hh + cr];
            }
            accnh[t] = v8f_zero();
        }

        // gh = h @ w_hh^T  (contraction over H=512)
        for (int kb = 0; kb < Hh; kb += 32) {
            v16bf a = load_a_frag(hs, HSTR, m, half, kb);
            #pragma unroll
            for (int t = 0; t < 4; ++t) {
                int cr = jb + 16 * t + m;
                v16bf b_r = load_b_frag(whh, Hh, cr,          half, kb);
                v16bf b_z = load_b_frag(whh, Hh, Hh + cr,     half, kb);
                v16bf b_n = load_b_frag(whh, Hh, 2 * Hh + cr, half, kb);
                accr[t]  = __builtin_amdgcn_wmma_f32_16x16x32_bf16(
                               false, a, false, b_r, (short)0, accr[t],  false, false);
                accz[t]  = __builtin_amdgcn_wmma_f32_16x16x32_bf16(
                               false, a, false, b_z, (short)0, accz[t],  false, false);
                accnh[t] = __builtin_amdgcn_wmma_f32_16x16x32_bf16(
                               false, a, false, b_n, (short)0, accnh[t], false, false);
            }
        }

        __syncthreads();  // every wave done reading hs

        #pragma unroll
        for (int t = 0; t < 4; ++t) {
            #pragma unroll
            for (int v = 0; v < 8; ++v) {
                float r    = fast_sigmoid(accr[t][v] + br[t]);
                float z    = fast_sigmoid(accz[t][v] + bz[t]);
                float n    = fast_tanh(accni[t][v] + r * (accnh[t][v] + bhn[t]));
                float hnew = (1.0f - z) * n + z * hf[t][v];
                hf[t][v] = hnew;
                int row = v + 8 * half;                 // D-layout: M = v + 8*half
                hs[row * HSTR + jb + 16 * t + m] = (bf16_t)hnew;
            }
        }
        __syncthreads();  // new h visible before next step
    }

    #pragma unroll
    for (int t = 0; t < 4; ++t) {
        #pragma unroll
        for (int v = 0; v < 8; ++v) {
            long long row = b0 + v + 8 * half;
            out[row * Hh + jb + 16 * t + m] = hf[t][v];
        }
    }
}

// =====================================================================
// Fallback: fully fused scan (used when workspace can't hold gi).
// =====================================================================
__launch_bounds__(256)
__global__ void gru_scan_fused_kernel(const float* __restrict__ tokens,
                                      const int* __restrict__ indices,
                                      const bf16_t* __restrict__ wih,
                                      const bf16_t* __restrict__ whh,
                                      const float* __restrict__ bih,
                                      const float* __restrict__ bhh,
                                      float* __restrict__ out) {
    __shared__ __align__(32) bf16_t xs[16 * XSTR];
    __shared__ __align__(32) bf16_t hs[16 * HSTR];

    const int tid  = threadIdx.x;
    const int lane = tid & 31;
    const int wave = tid >> 5;
    const int m    = lane & 15;
    const int half = lane >> 4;
    const int jb   = wave * 64;
    const long long b0 = (long long)blockIdx.x * 16;

    for (int i = tid; i < 16 * HSTR; i += 256) hs[i] = (bf16_t)0.0f;

    float br[4], bz[4], bin_[4], bhn[4];
    #pragma unroll
    for (int t = 0; t < 4; ++t) {
        int c = jb + 16 * t + m;
        br[t]   = bih[c]      + bhh[c];
        bz[t]   = bih[Hh + c] + bhh[Hh + c];
        bin_[t] = bih[2 * Hh + c];
        bhn[t]  = bhh[2 * Hh + c];
    }

    v8f hf[4];
    #pragma unroll
    for (int t = 0; t < 4; ++t) hf[t] = v8f_zero();

    __syncthreads();

    for (int k = 0; k < Kk; ++k) {
        for (int j = 0; j < 16; ++j) {
            int idx = indices[(b0 + j) * Kk + k];
            const float* src = tokens + ((b0 + j) * (long long)Nn + idx) * Dd;
            xs[j * XSTR + tid] = (bf16_t)src[tid];
        }
        __syncthreads();

        v8f accr[4], accz[4], accni[4], accnh[4];
        #pragma unroll
        for (int t = 0; t < 4; ++t) {
            accr[t] = v8f_zero(); accz[t] = v8f_zero();
            accni[t] = v8f_zero(); accnh[t] = v8f_zero();
        }

        for (int kb = 0; kb < Dd; kb += 32) {
            v16bf a = load_a_frag(xs, XSTR, m, half, kb);
            #pragma unroll
            for (int t = 0; t < 4; ++t) {
                int cr = jb + 16 * t + m;
                v16bf b_r = load_b_frag(wih, Dd, cr,          half, kb);
                v16bf b_z = load_b_frag(wih, Dd, Hh + cr,     half, kb);
                v16bf b_n = load_b_frag(wih, Dd, 2 * Hh + cr, half, kb);
                accr[t]  = __builtin_amdgcn_wmma_f32_16x16x32_bf16(
                               false, a, false, b_r, (short)0, accr[t],  false, false);
                accz[t]  = __builtin_amdgcn_wmma_f32_16x16x32_bf16(
                               false, a, false, b_z, (short)0, accz[t],  false, false);
                accni[t] = __builtin_amdgcn_wmma_f32_16x16x32_bf16(
                               false, a, false, b_n, (short)0, accni[t], false, false);
            }
        }

        for (int kb = 0; kb < Hh; kb += 32) {
            v16bf a = load_a_frag(hs, HSTR, m, half, kb);
            #pragma unroll
            for (int t = 0; t < 4; ++t) {
                int cr = jb + 16 * t + m;
                v16bf b_r = load_b_frag(whh, Hh, cr,          half, kb);
                v16bf b_z = load_b_frag(whh, Hh, Hh + cr,     half, kb);
                v16bf b_n = load_b_frag(whh, Hh, 2 * Hh + cr, half, kb);
                accr[t]  = __builtin_amdgcn_wmma_f32_16x16x32_bf16(
                               false, a, false, b_r, (short)0, accr[t],  false, false);
                accz[t]  = __builtin_amdgcn_wmma_f32_16x16x32_bf16(
                               false, a, false, b_z, (short)0, accz[t],  false, false);
                accnh[t] = __builtin_amdgcn_wmma_f32_16x16x32_bf16(
                               false, a, false, b_n, (short)0, accnh[t], false, false);
            }
        }

        __syncthreads();

        #pragma unroll
        for (int t = 0; t < 4; ++t) {
            #pragma unroll
            for (int v = 0; v < 8; ++v) {
                float r    = fast_sigmoid(accr[t][v] + br[t]);
                float z    = fast_sigmoid(accz[t][v] + bz[t]);
                float n    = fast_tanh(accni[t][v] + bin_[t] +
                                       r * (accnh[t][v] + bhn[t]));
                float hnew = (1.0f - z) * n + z * hf[t][v];
                hf[t][v] = hnew;
                int row = v + 8 * half;
                hs[row * HSTR + jb + 16 * t + m] = (bf16_t)hnew;
            }
        }
        __syncthreads();
    }

    #pragma unroll
    for (int t = 0; t < 4; ++t) {
        #pragma unroll
        for (int v = 0; v < 8; ++v) {
            long long row = b0 + v + 8 * half;
            out[row * Hh + jb + 16 * t + m] = hf[t][v];
        }
    }
}

extern "C" void kernel_launch(void* const* d_in, const int* in_sizes, int n_in,
                              void* d_out, int out_size, void* d_ws, size_t ws_size,
                              hipStream_t stream) {
    const float* tokens  = (const float*)d_in[0];
    const int*   indices = (const int*)  d_in[1];   // int32 on device (jax default)
    const float* w_ih    = (const float*)d_in[2];
    const float* w_hh    = (const float*)d_in[3];
    const float* b_ih    = (const float*)d_in[4];
    const float* b_hh    = (const float*)d_in[5];
    float*       out     = (float*)d_out;

    const size_t wih_elems = (size_t)G3H * Dd;       // 393216 bf16 = 768 KiB
    const size_t whh_elems = (size_t)G3H * Hh;       // 786432 bf16 = 1.5 MiB
    const size_t wbf_bytes = (wih_elems + whh_elems) * sizeof(bf16_t);
    const size_t gi_bytes  = (size_t)Bq * Kk * G3H * sizeof(float);   // 96 MiB

    bf16_t* wih_bf = (bf16_t*)d_ws;
    bf16_t* whh_bf = wih_bf + wih_elems;
    float*  gi     = (float*)((char*)d_ws + wbf_bytes);

    (void)in_sizes; (void)n_in; (void)out_size;

    cvt_weights_kernel<<<1024, 256, 0, stream>>>(w_ih, w_hh, wih_bf);

    if (ws_size >= wbf_bytes + gi_bytes) {
        // Phase 2: input projections at full-machine parallelism (1024 WGs).
        gi_gemm_kernel<<<Bq, 256, 0, stream>>>(tokens, indices, wih_bf, b_ih, gi);
        // Phase 3: lean serial scan (gh only, 2 barriers/step).
        gru_scan_pre_kernel<<<Bq / 16, 256, 0, stream>>>(gi, whh_bf, b_hh, out);
    } else {
        gru_scan_fused_kernel<<<Bq / 16, 256, 0, stream>>>(
            tokens, indices, wih_bf, whh_bf, b_ih, b_hh, out);
    }
}
